// PillarBackBone8xAIFAD_2877628088619
// MI455X (gfx1250) — matrix operand order repaired and checked by reading, not compiled, not run on hardware
//
#include <hip/hip_runtime.h>

typedef _Float16 f16;
typedef __attribute__((ext_vector_type(16))) _Float16 v16h;
typedef __attribute__((ext_vector_type(8)))  _Float16 v8h;
typedef __attribute__((ext_vector_type(8)))  float    v8f;

#define FB    1   // conv bias
#define FBN   2   // batchnorm
#define FRELU 4
#define FMASK 8
#define FRES  16  // residual add (pre-relu)

static inline long cdiv(long a, long b){ return (a + b - 1) / b; }

// ---------------------------------------------------------------- utilities
__global__ void fill_zero4(float4* p, long n){
  long i = (long)blockIdx.x * blockDim.x + threadIdx.x;
  if (i < n) p[i] = make_float4(0.f, 0.f, 0.f, 0.f);
}

__global__ void scatter_feats(const float* __restrict__ vf, const int* __restrict__ coords,
                              float* __restrict__ dense, int nvox, int D, int H, int W, int C){
  long i = (long)blockIdx.x * blockDim.x + threadIdx.x;
  if (i >= (long)nvox * C) return;
  int c = (int)(i % C); int n = (int)(i / C);
  int b = coords[n*4+0], z = coords[n*4+1], y = coords[n*4+2], x = coords[n*4+3];
  long idx = ((((long)b*D + z)*H + y)*(long)W + x)*C + c;
  atomicAdd(dense + idx, vf[i]);
}

__global__ void scatter_mask(const int* __restrict__ coords, f16* m, int nvox, int D, int H, int W){
  int n = blockIdx.x * blockDim.x + threadIdx.x;
  if (n >= nvox) return;
  int b = coords[n*4+0], z = coords[n*4+1], y = coords[n*4+2], x = coords[n*4+3];
  m[(((long)b*D + z)*H + y)*(long)W + x] = (f16)1.0f;
}

__global__ void f32_to_f16(const float* __restrict__ in, f16* __restrict__ out, long n){
  long i = (long)blockIdx.x * blockDim.x + threadIdx.x;
  if (i < n) out[i] = (f16)in[i];
}

// 3x3x3 mask max-pool, pad 1, stride s (also handles 2D case with Din=1,s=1)
__global__ void maxpool_mask(const f16* __restrict__ in, f16* __restrict__ out,
                             int B, int Din, int Hin, int Win,
                             int Dout, int Hout, int Wout, int s){
  long i = (long)blockIdx.x * blockDim.x + threadIdx.x;
  long total = (long)B * Dout * Hout * Wout;
  if (i >= total) return;
  int xo = (int)(i % Wout); long t = i / Wout;
  int yo = (int)(t % Hout); t /= Hout;
  int zo = (int)(t % Dout); int b = (int)(t / Dout);
  float m = 0.f;
  for (int dz=-1; dz<=1; ++dz){ int z = zo*s+dz; if ((unsigned)z >= (unsigned)Din) continue;
    for (int dy=-1; dy<=1; ++dy){ int y = yo*s+dy; if ((unsigned)y >= (unsigned)Hin) continue;
      for (int dx=-1; dx<=1; ++dx){ int x = xo*s+dx; if ((unsigned)x >= (unsigned)Win) continue;
        float v = (float)in[(((long)b*Din + z)*Hin + y)*(long)Win + x];
        m = fmaxf(m, v);
      }}}
  out[i] = (f16)m;
}

// f = x_hi + upsample2x(x_lo) placed at even coords (channels-last)
__global__ void fuse_combine(const f16* __restrict__ xhi, const f16* __restrict__ xlo,
                             f16* __restrict__ out, int B, int D, int H, int W,
                             int Dl, int Hl, int Wl, int C){
  long i = (long)blockIdx.x * blockDim.x + threadIdx.x;
  long total = (long)B * D * H * W * C;
  if (i >= total) return;
  int c = (int)(i % C); long t = i / C;
  int x = (int)(t % W); t /= W;
  int y = (int)(t % H); t /= H;
  int z = (int)(t % D); int b = (int)(t / D);
  float v = (float)xhi[i];
  if (!((z | y | x) & 1)){
    int zl = z >> 1, yl = y >> 1, xl = x >> 1;
    if (zl < Dl && yl < Hl && xl < Wl)
      v += (float)xlo[((((long)b*Dl + zl)*Hl + yl)*(long)Wl + xl)*C + c];
  }
  out[i] = (f16)v;
}

__global__ void fuse_mask_k(const f16* __restrict__ mhi, const f16* __restrict__ mlo,
                            f16* __restrict__ out, int B, int D, int H, int W,
                            int Dl, int Hl, int Wl){
  long i = (long)blockIdx.x * blockDim.x + threadIdx.x;
  long total = (long)B * D * H * W;
  if (i >= total) return;
  int x = (int)(i % W); long t = i / W;
  int y = (int)(t % H); t /= H;
  int z = (int)(t % D); int b = (int)(t / D);
  float v = (float)mhi[i];
  if (!((z | y | x) & 1)){
    int zl = z >> 1, yl = y >> 1, xl = x >> 1;
    if (zl < Dl && yl < Hl && xl < Wl)
      v = fmaxf(v, (float)mlo[(((long)b*Dl + zl)*Hl + yl)*(long)Wl + xl]);
  }
  out[i] = (f16)v;
}

// Repack OIDHW (or OIHW with is2d) fp32 weights into exact WMMA B-fragment order:
// out[ctile][kk][lane][16] f16, N=lane%16, K=8*(lane/16)+(e&7)+16*(e/8)
__global__ void repack_w(const float* __restrict__ w, f16* __restrict__ out,
                         int Cout, int Cin, int KK, int KredReal, int is2d){
  long i = (long)blockIdx.x * blockDim.x + threadIdx.x;
  long total = (long)(Cout/16) * KK * 512;
  if (i >= total) return;
  int e = (int)(i & 15); int lane = (int)((i >> 4) & 31); long t = i >> 9;
  int kk = (int)(t % KK); int ctile = (int)(t / KK);
  int N = lane & 15; int K = (e & 7) + 8*(lane >> 4) + 16*(e >> 3);
  int gk = kk*32 + K; int cout = ctile*16 + N;
  float v = 0.f;
  if (gk < KredReal){
    int tap = gk / Cin, cin = gk - tap*Cin;
    int kd = tap/9, r = tap%9, kh = r/3, kw = r%3;
    if (is2d){ if (kd == 1) v = w[(((long)cout*Cin + cin)*3 + kh)*3 + kw]; }
    else       v = w[((((long)cout*Cin + cin)*3 + kd)*3 + kh)*3 + kw];
  }
  out[i] = (f16)v;
}

__global__ void to_nchw(const f16* __restrict__ in, float* __restrict__ out,
                        int B, int H, int W, int C){
  long i = (long)blockIdx.x * blockDim.x + threadIdx.x;
  long total = (long)B * C * H * W;
  if (i >= total) return;
  int x = (int)(i % W); long t = i / W;
  int y = (int)(t % H); t /= H;
  int c = (int)(t % C); int b = (int)(t / C);
  out[i] = (float)in[(((long)b*H + y)*(long)W + x)*C + c];
}

// ---------------------------------------------------------------- conv (implicit GEMM, WMMA f16)
struct ConvArgs {
  const f16* in; const f16* wfrag;
  const float* bias; const float* gamma; const float* beta;
  const f16* mask; const f16* resid;
  f16* out;
  int B, Din, Hin, Win, Cin, Dout, Hout, Wout, Cout, stride, KK, KredReal, flags, WT, tiles;
  int cinShift, cinMask;   // Cin is a power of two: replace div/mod with shift/mask
};

// One 8-channel contiguous run of the A fragment (zero for padding / OOB taps).
__device__ __forceinline__ v8h load_run(const f16* __restrict__ in, int b,
    int Din, int Hin, int Win, int Cin, int cinShift, int cinMask,
    int zi0, int yi0, int xi0, int gk, int KredReal){
  v8h z = {};
  if (gk >= KredReal) return z;
  int tap = gk >> cinShift;       // gk / Cin  (Cin = 2^cinShift)
  int c   = gk & cinMask;         // gk % Cin  (run of 8 stays inside one tap)
  int zi = zi0 + tap / 9;         // constant-divisor: cheap
  int r  = tap % 9;
  int yi = yi0 + r / 3;
  int xi = xi0 + r % 3;
  if ((unsigned)zi >= (unsigned)Din) return z;
  if ((unsigned)yi >= (unsigned)Hin) return z;
  if ((unsigned)xi >= (unsigned)Win) return z;
  long idx = ((((long)b*Din + zi)*Hin + yi)*(long)Win + xi)*Cin + c;
  return *(const v8h*)(in + idx);   // 16B aligned: Cin,c multiples of 8
}

// Block = 4 waves; each wave computes 16 consecutive W-pixels x 16 out-channels.
// All waves share blockIdx.y's 16-out-channel weight panel, staged once in LDS via
// CDNA5 async global->LDS DMA (GLOBAL_LOAD_ASYNC_TO_LDS_B128, ASYNCcnt-tracked).
template<int KKCAP>
__global__ __launch_bounds__(128) void conv_wmma(ConvArgs p){
  __shared__ __align__(32) f16 ldsW[KKCAP * 512];
  {
    const f16* __restrict__ src = p.wfrag + (size_t)blockIdx.y * ((size_t)p.KK * 512);
    int nW = p.KK * 512;
    for (int i = threadIdx.x * 8; i < nW; i += 128 * 8){
      unsigned lo = (unsigned)(size_t)(const void*)(ldsW + i);   // 0-based LDS byte offset
      unsigned long long ga = (unsigned long long)(size_t)(const void*)(src + i);
      __builtin_prefetch(src + i + 1024, 0, 0);                  // global_prefetch_b8
      asm volatile("global_load_async_to_lds_b128 %0, %1, off"   // 16B/lane mem->LDS DMA
                   :: "v"(lo), "v"(ga) : "memory");
    }
    asm volatile("s_wait_asynccnt 0x0" ::: "memory");            // drain ASYNCcnt
  }
  __syncthreads();

  const int wave = threadIdx.x >> 5;
  const int lane = threadIdx.x & 31;
  const int half = lane >> 4;
  const int l15  = lane & 15;
  int tile = blockIdx.x * 4 + wave;                      // wave-uniform
  if (tile >= p.tiles) return;

  int wt  = tile % p.WT;
  int row = tile / p.WT;
  int ho = row % p.Hout; int t2 = row / p.Hout;
  int dd = t2 % p.Dout;  int b  = t2 / p.Dout;
  int x0 = wt << 4;
  int zi0 = dd * p.stride - 1;
  int yi0 = ho * p.stride - 1;
  int xi0 = (x0 + l15) * p.stride - 1;                   // A row M = l15

  v8f acc = {};
  const f16* lw = ldsW + lane * 16;
#pragma unroll 2
  for (int kk = 0; kk < p.KK; ++kk){
    int gk0 = (kk << 5) + (half << 3);
    v8h r0 = load_run(p.in, b, p.Din, p.Hin, p.Win, p.Cin, p.cinShift, p.cinMask,
                      zi0, yi0, xi0, gk0,      p.KredReal);
    v8h r1 = load_run(p.in, b, p.Din, p.Hin, p.Win, p.Cin, p.cinShift, p.cinMask,
                      zi0, yi0, xi0, gk0 + 16, p.KredReal);
    v16h amat;
#pragma unroll
    for (int e = 0; e < 8; ++e){ amat[e] = r0[e]; amat[e + 8] = r1[e]; }
    v16h bmat = *(const v16h*)(lw + (kk << 9));          // ds_load_b128 x2
    acc = __builtin_amdgcn_wmma_f32_16x16x32_f16(false, amat, false, bmat,
                                                 (short)0, acc, false, false);
  }

  // Fused epilogue: +bias, BN, +residual, ReLU, *mask.  C/D: M=j+8*half, N=l15.
  int n = blockIdx.y * 16 + l15;
  float scale = 1.f, shift = 0.f, cb = 0.f;
  if (p.flags & FBN){ scale = p.gamma[n] * 0.999500374f; shift = p.beta[n]; } // 1/sqrt(1+1e-3)
  if (p.flags & FB)  cb = p.bias[n];
  long obase = (((long)b * p.Dout + dd) * p.Hout + ho) * (long)p.Wout;
#pragma unroll
  for (int j = 0; j < 8; ++j){
    int xw = x0 + (half << 3) + j;
    if (xw >= p.Wout) continue;
    long pix = obase + xw;
    float v = acc[j] + cb;
    if (p.flags & FBN)   v = v * scale + shift;
    if (p.flags & FRES)  v += (float)p.resid[pix * p.Cout + n];
    if (p.flags & FRELU) v = fmaxf(v, 0.f);
    if (p.flags & FMASK) v *= (float)p.mask[pix];
    p.out[pix * p.Cout + n] = (f16)v;
  }
}

static void run_conv(hipStream_t stream, f16* FR,
                     const f16* in, const float* w, const f16* mask, const f16* resid,
                     const float* bias, const float* g, const float* be, f16* out,
                     int B, int Din, int Hin, int Win, int Cin, int Cout,
                     int s, int is2d, int flags){
  int KredReal = 27 * Cin;
  int KK = (int)cdiv(KredReal, 32);
  long rtot = (long)(Cout/16) * KK * 512;
  repack_w<<<dim3((unsigned)cdiv(rtot, 256)), 256, 0, stream>>>(w, FR, Cout, Cin, KK, KredReal, is2d);
  ConvArgs a;
  a.in = in; a.wfrag = FR; a.bias = bias; a.gamma = g; a.beta = be;
  a.mask = mask; a.resid = resid; a.out = out;
  a.B = B; a.Din = Din; a.Hin = Hin; a.Win = Win; a.Cin = Cin;
  a.Dout = (s == 2) ? ((Din - 1) / 2 + 1) : Din;
  a.Hout = (s == 2) ? ((Hin - 1) / 2 + 1) : Hin;
  a.Wout = (s == 2) ? ((Win - 1) / 2 + 1) : Win;
  a.Cout = Cout; a.stride = s; a.KK = KK; a.KredReal = KredReal; a.flags = flags;
  a.WT = (a.Wout + 15) / 16;
  a.tiles = a.B * a.Dout * a.Hout * a.WT;
  a.cinShift = __builtin_ctz((unsigned)Cin);
  a.cinMask  = Cin - 1;
  dim3 grid((unsigned)cdiv(a.tiles, 4), (unsigned)(Cout / 16));
  if      (KK <= 14) conv_wmma<14><<<grid, 128, 0, stream>>>(a);
  else if (KK <= 27) conv_wmma<27><<<grid, 128, 0, stream>>>(a);
  else if (KK <= 54) conv_wmma<54><<<grid, 128, 0, stream>>>(a);
  else               conv_wmma<108><<<grid, 128, 0, stream>>>(a);
}

// ---------------------------------------------------------------- orchestration
extern "C" void kernel_launch(void* const* d_in, const int* in_sizes, int n_in,
                              void* d_out, int out_size, void* d_ws, size_t ws_size,
                              hipStream_t stream) {
  (void)n_in; (void)out_size; (void)ws_size;
  const float* vf     = (const float*)d_in[0];
  const int*   coords = (const int*)d_in[1];
  const int nvox = in_sizes[1] / 4;

  int idx = 2;
  auto P = [&]() -> const float* { return (const float*)d_in[idx++]; };
  struct DownP { const float *w, *g, *be; };
  struct ResP  { const float *w1, *b1, *g1, *be1, *w2, *b2, *g2, *be2; };
  struct FuseP { const float *w, *b, *g, *be; };
  auto getDown = [&](){ DownP p; p.w=P(); p.g=P(); p.be=P(); return p; };
  auto getRes  = [&](){ ResP p; p.w1=P(); p.b1=P(); p.g1=P(); p.be1=P();
                        p.w2=P(); p.b2=P(); p.g2=P(); p.be2=P(); return p; };
  auto getFuse = [&](){ FuseP p; p.w=P(); p.b=P(); p.g=P(); p.be=P(); return p; };

  const float* w_in = P(); const float* g_in = P(); const float* be_in = P();
  DownP down1 = getDown(); ResP res1 = getRes(); FuseP fuse1 = getFuse();
  DownP down2 = getDown(); ResP res2 = getRes();
  DownP down22 = getDown(); ResP res22 = getRes(); FuseP fuse2 = getFuse();
  DownP down3 = getDown(); ResP res3 = getRes();
  DownP down32 = getDown(); ResP res32 = getRes(); FuseP fuse3 = getFuse();
  DownP down4 = getDown(); ResP res4 = getRes();
  DownP down42 = getDown(); ResP res42 = getRes(); FuseP fuse4 = getFuse();
  const float* w_out = P(); const float* g_out = P(); const float* be_out = P();
  const float* w_sh = P(); const float* b_sh = P(); const float* g_sh = P(); const float* be_sh = P();

  const int Bc = 2;
  const int D0 = 2, H0 = 560, W0 = 560; const long NB0 = (long)Bc*D0*H0*W0;
  const int D1 = 1, H1 = 280, W1 = 280; const long NB1 = (long)Bc*D1*H1*W1;
  const int D2 = 1, H2 = 140, W2 = 140; const long NB2 = (long)Bc*D2*H2*W2;
  const int D3 = 1, H3 = 70,  W3 = 70;  const long NB3 = (long)Bc*D3*H3*W3;
  const int D4 = 1, H4 = 35,  W4 = 35;  const long NB4 = (long)Bc*D4*H4*W4;

  char* ws = (char*)d_ws; size_t off = 0;
  auto alloc = [&](size_t bytes) -> void* {
    void* p = ws + off; off = (off + bytes + 255) & ~(size_t)255; return p;
  };

  // big scatter buffer; region reused for h / f1tmp / f1 after f32->f16 convert
  float* dense32 = (float*)alloc(NB0 * 32 * sizeof(float));
  f16* h   = (f16*)dense32;
  f16* f1t = h   + (size_t)NB0 * 16;
  f16* f1  = f1t + (size_t)NB0 * 16;     // 3*NB0*16*2 = 120.4MB <= 160.6MB

  f16* dense16 = (f16*)alloc(NB0 * 32 * 2);
  f16* m0  = (f16*)alloc(NB0 * 2);
  f16* fm1 = (f16*)alloc(NB0 * 2);
  f16* m1  = (f16*)alloc(NB1 * 2);
  f16* d1  = (f16*)alloc(NB1 * 16 * 2);
  f16* rT  = (f16*)alloc(NB1 * 32 * 2);  // shared residual temp (max size)
  f16* x1  = (f16*)alloc(NB1 * 16 * 2);
  f16* m2  = (f16*)alloc(NB1 * 2);
  f16* d2  = (f16*)alloc(NB1 * 32 * 2);
  f16* x2  = (f16*)alloc(NB1 * 32 * 2);
  f16* m22 = (f16*)alloc(NB2 * 2);
  f16* d22 = (f16*)alloc(NB2 * 32 * 2);
  f16* x22 = (f16*)alloc(NB2 * 32 * 2);
  f16* fm2 = (f16*)alloc(NB1 * 2);
  f16* f2t = (f16*)alloc(NB1 * 32 * 2);
  f16* f2  = (f16*)alloc(NB1 * 32 * 2);
  f16* m3  = (f16*)alloc(NB2 * 2);
  f16* d3  = (f16*)alloc(NB2 * 64 * 2);
  f16* x3  = (f16*)alloc(NB2 * 64 * 2);
  f16* m32 = (f16*)alloc(NB3 * 2);
  f16* d32 = (f16*)alloc(NB3 * 64 * 2);
  f16* x32 = (f16*)alloc(NB3 * 64 * 2);
  f16* fm3 = (f16*)alloc(NB2 * 2);
  f16* f3t = (f16*)alloc(NB2 * 64 * 2);
  f16* f3  = (f16*)alloc(NB2 * 64 * 2);
  f16* m4  = (f16*)alloc(NB3 * 2);
  f16* d4  = (f16*)alloc(NB3 * 128 * 2);
  f16* x4  = (f16*)alloc(NB3 * 128 * 2);
  f16* m42 = (f16*)alloc(NB4 * 2);
  f16* d42 = (f16*)alloc(NB4 * 128 * 2);
  f16* x42 = (f16*)alloc(NB4 * 128 * 2);
  f16* fm4 = (f16*)alloc(NB3 * 2);
  f16* f4t = (f16*)alloc(NB3 * 128 * 2);
  f16* f4  = (f16*)alloc(NB3 * 128 * 2);
  f16* mo  = (f16*)alloc(NB3 * 2);
  f16* o1  = (f16*)alloc(NB3 * 128 * 2);
  f16* o2  = (f16*)alloc(NB3 * 128 * 2);
  f16* FR  = (f16*)alloc((size_t)8 * 108 * 512 * 2);   // shared weight-fragment buffer

  auto zero = [&](void* p, long bytes){
    long n4 = bytes / 16;
    fill_zero4<<<dim3((unsigned)cdiv(n4, 256)), 256, 0, stream>>>((float4*)p, n4);
  };
  auto pool = [&](const f16* mi, f16* mo_, int Din_, int Hin_, int Win_, int s){
    int Do = (s==2)?((Din_-1)/2+1):Din_, Ho = (s==2)?((Hin_-1)/2+1):Hin_, Wo = (s==2)?((Win_-1)/2+1):Win_;
    long tot = (long)Bc * Do * Ho * Wo;
    maxpool_mask<<<dim3((unsigned)cdiv(tot,256)), 256, 0, stream>>>(mi, mo_, Bc, Din_, Hin_, Win_, Do, Ho, Wo, s);
  };
  auto fuse_c = [&](const f16* xhi, const f16* xlo, f16* fo,
                    int D, int H, int W, int Dl, int Hl, int Wl, int C){
    long tot = (long)Bc * D * H * W * C;
    fuse_combine<<<dim3((unsigned)cdiv(tot,256)), 256, 0, stream>>>(xhi, xlo, fo, Bc, D, H, W, Dl, Hl, Wl, C);
  };
  auto fuse_m = [&](const f16* mhi, const f16* mlo, f16* mout,
                    int D, int H, int W, int Dl, int Hl, int Wl){
    long tot = (long)Bc * D * H * W;
    fuse_mask_k<<<dim3((unsigned)cdiv(tot,256)), 256, 0, stream>>>(mhi, mlo, mout, Bc, D, H, W, Dl, Hl, Wl);
  };

  // ---- scatter to dense grid (channels-last), build mask
  zero(dense32, NB0 * 32 * 4);
  zero(m0, NB0 * 2);
  scatter_feats<<<dim3((unsigned)cdiv((long)nvox*32, 256)), 256, 0, stream>>>(vf, coords, dense32, nvox, D0, H0, W0, 32);
  scatter_mask<<<dim3((unsigned)cdiv(nvox, 256)), 256, 0, stream>>>(coords, m0, nvox, D0, H0, W0);
  f32_to_f16<<<dim3((unsigned)cdiv(NB0*32, 256)), 256, 0, stream>>>(dense32, dense16, NB0 * 32);

  // ---- stem
  run_conv(stream, FR, dense16, w_in, m0, nullptr, nullptr, g_in, be_in, h,
           Bc, D0, H0, W0, 32, 16, 1, 0, FBN|FRELU|FMASK);

  // ---- level 1
  pool(m0, m1, D0, H0, W0, 2);
  run_conv(stream, FR, h, down1.w, m1, nullptr, nullptr, down1.g, down1.be, d1,
           Bc, D0, H0, W0, 16, 16, 2, 0, FBN|FRELU|FMASK);
  run_conv(stream, FR, d1, res1.w1, m1, nullptr, res1.b1, res1.g1, res1.be1, rT,
           Bc, D1, H1, W1, 16, 16, 1, 0, FB|FBN|FRELU|FMASK);
  run_conv(stream, FR, rT, res1.w2, m1, d1, res1.b2, res1.g2, res1.be2, x1,
           Bc, D1, H1, W1, 16, 16, 1, 0, FB|FBN|FRES|FRELU|FMASK);
  fuse_c(h, x1, f1t, D0, H0, W0, D1, H1, W1, 16);
  fuse_m(m0, m1, fm1, D0, H0, W0, D1, H1, W1);
  run_conv(stream, FR, f1t, fuse1.w, fm1, nullptr, fuse1.b, fuse1.g, fuse1.be, f1,
           Bc, D0, H0, W0, 16, 16, 1, 0, FB|FBN|FRELU|FMASK);

  // ---- level 2
  pool(fm1, m2, D0, H0, W0, 2);
  run_conv(stream, FR, f1, down2.w, m2, nullptr, nullptr, down2.g, down2.be, d2,
           Bc, D0, H0, W0, 16, 32, 2, 0, FBN|FRELU|FMASK);
  run_conv(stream, FR, d2, res2.w1, m2, nullptr, res2.b1, res2.g1, res2.be1, rT,
           Bc, D1, H1, W1, 32, 32, 1, 0, FB|FBN|FRELU|FMASK);
  run_conv(stream, FR, rT, res2.w2, m2, d2, res2.b2, res2.g2, res2.be2, x2,
           Bc, D1, H1, W1, 32, 32, 1, 0, FB|FBN|FRES|FRELU|FMASK);
  pool(m2, m22, D1, H1, W1, 2);
  run_conv(stream, FR, x2, down22.w, m22, nullptr, nullptr, down22.g, down22.be, d22,
           Bc, D1, H1, W1, 32, 32, 2, 0, FBN|FRELU|FMASK);
  run_conv(stream, FR, d22, res22.w1, m22, nullptr, res22.b1, res22.g1, res22.be1, rT,
           Bc, D2, H2, W2, 32, 32, 1, 0, FB|FBN|FRELU|FMASK);
  run_conv(stream, FR, rT, res22.w2, m22, d22, res22.b2, res22.g2, res22.be2, x22,
           Bc, D2, H2, W2, 32, 32, 1, 0, FB|FBN|FRES|FRELU|FMASK);
  fuse_c(x2, x22, f2t, D1, H1, W1, D2, H2, W2, 32);
  fuse_m(m2, m22, fm2, D1, H1, W1, D2, H2, W2);
  run_conv(stream, FR, f2t, fuse2.w, fm2, nullptr, fuse2.b, fuse2.g, fuse2.be, f2,
           Bc, D1, H1, W1, 32, 32, 1, 0, FB|FBN|FRELU|FMASK);

  // ---- level 3
  pool(fm2, m3, D1, H1, W1, 2);
  run_conv(stream, FR, f2, down3.w, m3, nullptr, nullptr, down3.g, down3.be, d3,
           Bc, D1, H1, W1, 32, 64, 2, 0, FBN|FRELU|FMASK);
  run_conv(stream, FR, d3, res3.w1, m3, nullptr, res3.b1, res3.g1, res3.be1, rT,
           Bc, D2, H2, W2, 64, 64, 1, 0, FB|FBN|FRELU|FMASK);
  run_conv(stream, FR, rT, res3.w2, m3, d3, res3.b2, res3.g2, res3.be2, x3,
           Bc, D2, H2, W2, 64, 64, 1, 0, FB|FBN|FRES|FRELU|FMASK);
  pool(m3, m32, D2, H2, W2, 2);
  run_conv(stream, FR, x3, down32.w, m32, nullptr, nullptr, down32.g, down32.be, d32,
           Bc, D2, H2, W2, 64, 64, 2, 0, FBN|FRELU|FMASK);
  run_conv(stream, FR, d32, res32.w1, m32, nullptr, res32.b1, res32.g1, res32.be1, rT,
           Bc, D3, H3, W3, 64, 64, 1, 0, FB|FBN|FRELU|FMASK);
  run_conv(stream, FR, rT, res32.w2, m32, d32, res32.b2, res32.g2, res32.be2, x32,
           Bc, D3, H3, W3, 64, 64, 1, 0, FB|FBN|FRES|FRELU|FMASK);
  fuse_c(x3, x32, f3t, D2, H2, W2, D3, H3, W3, 64);
  fuse_m(m3, m32, fm3, D2, H2, W2, D3, H3, W3);
  run_conv(stream, FR, f3t, fuse3.w, fm3, nullptr, fuse3.b, fuse3.g, fuse3.be, f3,
           Bc, D2, H2, W2, 64, 64, 1, 0, FB|FBN|FRELU|FMASK);

  // ---- level 4
  pool(fm3, m4, D2, H2, W2, 2);
  run_conv(stream, FR, f3, down4.w, m4, nullptr, nullptr, down4.g, down4.be, d4,
           Bc, D2, H2, W2, 64, 128, 2, 0, FBN|FRELU|FMASK);
  run_conv(stream, FR, d4, res4.w1, m4, nullptr, res4.b1, res4.g1, res4.be1, rT,
           Bc, D3, H3, W3, 128, 128, 1, 0, FB|FBN|FRELU|FMASK);
  run_conv(stream, FR, rT, res4.w2, m4, d4, res4.b2, res4.g2, res4.be2, x4,
           Bc, D3, H3, W3, 128, 128, 1, 0, FB|FBN|FRES|FRELU|FMASK);
  pool(m4, m42, D3, H3, W3, 2);
  run_conv(stream, FR, x4, down42.w, m42, nullptr, nullptr, down42.g, down42.be, d42,
           Bc, D3, H3, W3, 128, 128, 2, 0, FBN|FRELU|FMASK);
  run_conv(stream, FR, d42, res42.w1, m42, nullptr, res42.b1, res42.g1, res42.be1, rT,
           Bc, D4, H4, W4, 128, 128, 1, 0, FB|FBN|FRELU|FMASK);
  run_conv(stream, FR, rT, res42.w2, m42, d42, res42.b2, res42.g2, res42.be2, x42,
           Bc, D4, H4, W4, 128, 128, 1, 0, FB|FBN|FRES|FRELU|FMASK);
  fuse_c(x4, x42, f4t, D3, H3, W3, D4, H4, W4, 128);
  fuse_m(m4, m42, fm4, D3, H3, W3, D4, H4, W4);
  run_conv(stream, FR, f4t, fuse4.w, fm4, nullptr, fuse4.b, fuse4.g, fuse4.be, f4,
           Bc, D3, H3, W3, 128, 128, 1, 0, FB|FBN|FRELU|FMASK);

  // ---- BEV head (D3==1: f4 is already the BEV map; mb == fm4)
  pool(fm4, mo, 1, H3, W3, 1);   // 2D 3x3 stride-1 max-pool (depth window collapses)
  run_conv(stream, FR, f4, w_out, mo, nullptr, nullptr, g_out, be_out, o1,
           Bc, 1, H3, W3, 128, 128, 1, 1, FBN|FRELU|FMASK);
  run_conv(stream, FR, o1, w_sh, mo, nullptr, b_sh, g_sh, be_sh, o2,
           Bc, 1, H3, W3, 128, 128, 1, 1, FB|FBN|FRELU|FMASK);

  long tot = (long)Bc * 128 * H3 * W3;
  to_nchw<<<dim3((unsigned)cdiv(tot, 256)), 256, 0, stream>>>(o2, (float*)d_out, Bc, H3, W3, 128);
}
